// SwinTransformerBlock_16698832847104
// MI455X (gfx1250) — matrix operand order repaired and checked by reading, not compiled, not run on hardware
//
#include <hip/hip_runtime.h>

typedef __attribute__((ext_vector_type(16))) __bf16 bf16x16;
typedef __attribute__((ext_vector_type(8)))  float  f32x8;
typedef __attribute__((ext_vector_type(4)))  float  f32x4;
typedef __attribute__((ext_vector_type(4)))  unsigned int uint4v;

#define DEV static __device__ __forceinline__

constexpr int BATCH = 32, IH = 56, IW = 56, DIMC = 256, NHEADS = 8, WS = 7, SHIFT = 3;
constexpr int NWIN   = (IH / WS) * (IW / WS);  // 64 windows per image
constexpr int HID    = 4 * DIMC;               // 1024
constexpr float ATT_SCALE = 0.17677669529663687f;  // (256/8)^-0.5
constexpr float LNEPS = 1e-5f;

// ---------- scalar helpers ----------
// fp32 -> bf16 via native hardware convert (RNE), avoids software rounding sequences.
DEV unsigned short f2bf(float f) {
  union { __bf16 b; unsigned short u; } c;
  c.b = (__bf16)f;
  return c.u;
}
DEV f32x8 zero8() {
  f32x8 z;
#pragma unroll
  for (int i = 0; i < 8; ++i) z[i] = 0.f;
  return z;
}
DEV float wave_max(float v) {
#pragma unroll
  for (int off = 16; off > 0; off >>= 1) v = fmaxf(v, __shfl_xor(v, off, 32));
  return v;
}
DEV float wave_sum(float v) {
#pragma unroll
  for (int off = 16; off > 0; off >>= 1) v += __shfl_xor(v, off, 32);
  return v;
}

union frag_cast { uint4v u[2]; bf16x16 v; };

// ---------- WMMA fragment loaders (CDNA5 wave32 layouts, 16-bit data) ----------
// A fragment 16x32 from row-major [row][k]. Per lane the 16 elements are one
// contiguous 16-element K-run: K = khi+0..15  ->  2x b128 loads.
DEV bf16x16 frag_a(const unsigned short* src, int row_base, int k_base, int ld, int lane) {
  const int row = row_base + (lane & 15);
  const int khi = (lane >= 16) ? 16 : 0;
  const unsigned short* p = src + row * ld + k_base + khi;
  frag_cast f;
  f.u[0] = *(const uint4v*)p;         // elements 0..7   (K = khi+0..7)
  f.u[1] = *(const uint4v*)(p + 8);   // elements 8..15  (K = khi+8..15)
  return f.v;
}
// B fragment 32x16 where logical B[k][n] = src[n][k] (row-major [n][k], i.e. a
// transposed operand). Per lane: N fixed, elements 0..15 <-> K = khi+0..15
// contiguous -> 2x b128 loads. Used for K^T (scores) and V^T (P.V).
DEV bf16x16 frag_bt(const unsigned short* src, int k_base, int n_base, int ld, int lane) {
  const int n = n_base + (lane & 15);
  const int khi = (lane >= 16) ? 16 : 0;
  const unsigned short* p = src + n * ld + k_base + khi;
  frag_cast f;
  f.u[0] = *(const uint4v*)p;
  f.u[1] = *(const uint4v*)(p + 8);
  return f.v;
}
// B fragment from fragment-major pre-swizzled weights: each lane's 16 elements
// stored contiguously (32 B/lane, 1 KB/tile) -> 2x global_load_b128, coalesced.
DEV bf16x16 frag_b_swz(const unsigned short* base, int tile_id, int lane) {
  const unsigned short* p = base + ((tile_id * 32 + lane) << 4);
  frag_cast f;
  f.u[0] = *(const uint4v*)p;
  f.u[1] = *(const uint4v*)(p + 8);
  return f.v;
}
DEV f32x8 wmma_bf16(bf16x16 a, bf16x16 b, f32x8 c) {
  return __builtin_amdgcn_wmma_f32_16x16x32_bf16(false, a, false, b, (short)0, c, false, false);
}

// ---------- weight fp32 -> bf16 conversion with fragment-major swizzle ----------
// dst element i = ((tile*32 + lane)*16 + e); tile = kb*(N/16) + nt;
// source (k, n) = (kb*32 + (lane>=16?16:0) + e, nt*16 + (lane&15)).
__global__ void cvt_swz(const float* __restrict__ src, unsigned short* __restrict__ dst,
                        int K, int N) {
  const int i = blockIdx.x * 256 + threadIdx.x;
  if (i >= K * N) return;
  const int tile = i >> 9, r = i & 511, lane = r >> 4, e = r & 15;
  const int ntiles = N >> 4;
  const int kb = tile / ntiles, nt = tile - kb * ntiles;
  const int k = kb * 32 + ((lane >= 16) ? 16 : 0) + e;
  const int n = nt * 16 + (lane & 15);
  dst[i] = f2bf(src[k * N + n]);
}

// =====================================================================================
// Fused per-window kernel: LN1 + shift-gather + QKV GEMMs + masked attention + softmax
// + P.V + output projection + window-reverse/unshift + residual. One block per window,
// 8 waves (wave32). Entire 64x256 working set lives in LDS (160 KB of the 320 KB WGP LDS).
// =====================================================================================
__global__ __launch_bounds__(256) void swin_attn_fused(
    const float* __restrict__ x, const float* __restrict__ mask,
    const float* __restrict__ n1g, const float* __restrict__ n1b,
    const unsigned short* __restrict__ wq, const unsigned short* __restrict__ wk,
    const unsigned short* __restrict__ wv, const float* __restrict__ btab,
    const int* __restrict__ relidx, const unsigned short* __restrict__ wproj,
    const float* __restrict__ projb, float* __restrict__ x1) {
  __shared__ unsigned short s_x[64 * 256];   // LN'd window (bf16); reused for attn output
  __shared__ unsigned short s_q[64 * 256];   // Q   [token][chan]
  __shared__ unsigned short s_k[64 * 256];   // K   [token][chan]
  __shared__ unsigned short s_v[256 * 64];   // V^T [chan][token]  (transposed at store)
  __shared__ float s_sc[8 * 16 * 64];        // per-wave score staging; probs (bf16) overlay

  const int tid = threadIdx.x, lane = tid & 31, wid = tid >> 5;
  const int win = blockIdx.x;
  const int bimg = win >> 6, w_in = win & 63;
  const int wh = w_in >> 3, ww = w_in & 7;
  const int hi8 = (lane >= 16) ? 8 : 0;

  // ---- Phase 0: shift-gather + LayerNorm1 -> bf16 LDS (rows 49..63 zero-padded) ----
  for (int r = wid; r < 64; r += 8) {
    if (r < WS * WS) {
      const int i = r / WS, j = r % WS;
      const int ho = (wh * WS + i + SHIFT) % IH;
      const int wo = (ww * WS + j + SHIFT) % IW;
      const float* src = x + ((size_t)bimg * IH * IW + ho * IW + wo) * DIMC;
      const f32x4 v0 = ((const f32x4*)(src + lane * 8))[0];
      const f32x4 v1 = ((const f32x4*)(src + lane * 8))[1];
      float v[8] = {v0.x, v0.y, v0.z, v0.w, v1.x, v1.y, v1.z, v1.w};
      float s = 0.f, ss = 0.f;
#pragma unroll
      for (int u = 0; u < 8; ++u) { s += v[u]; ss += v[u] * v[u]; }
      s = wave_sum(s); ss = wave_sum(ss);
      const float mu = s * (1.f / DIMC);
      const float rs = rsqrtf(ss * (1.f / DIMC) - mu * mu + LNEPS);
#pragma unroll
      for (int u = 0; u < 8; ++u) {
        const int c = lane * 8 + u;
        s_x[r * DIMC + c] = f2bf((v[u] - mu) * rs * n1g[c] + n1b[c]);
      }
    } else {
#pragma unroll
      for (int u = 0; u < 8; ++u) s_x[r * DIMC + lane * 8 + u] = 0;
    }
  }
  __syncthreads();

  // ---- Phase 1: Q/K/V GEMMs (64x256 @ 256x256, bf16 WMMA, f32 accum) ----
  for (int t = wid; t < 3 * 64; t += 8) {
    const int which = t >> 6, tt = t & 63;
    const int m = tt >> 4, n = tt & 15;
    const unsigned short* wsel = (which == 0) ? wq : (which == 1) ? wk : wv;
    f32x8 acc = zero8();
#pragma unroll
    for (int kb = 0; kb < 8; ++kb) {
      bf16x16 a = frag_a(s_x, m * 16, kb * 32, DIMC, lane);
      bf16x16 b = frag_b_swz(wsel, kb * 16 + n, lane);  // N=256 -> 16 tiles per K-block
      acc = wmma_bf16(a, b, acc);
    }
    if (which == 2) {  // V stored transposed: V^T[chan][token]
      const int col = n * 16 + (lane & 15);
#pragma unroll
      for (int r = 0; r < 8; ++r)
        s_v[col * 64 + m * 16 + r + hi8] = f2bf(acc[r]);
    } else {
      unsigned short* dsel = (which == 0) ? s_q : s_k;
#pragma unroll
      for (int r = 0; r < 8; ++r)
        dsel[(m * 16 + r + hi8) * DIMC + n * 16 + (lane & 15)] = f2bf(acc[r]);
    }
  }
  __syncthreads();

  // ---- Phase 2: attention, one head per wave ----
  {
    const int head = wid;
    float* sc = s_sc + wid * 16 * 64;
    unsigned short* psc = (unsigned short*)sc;  // probs overlay (row stride 128 u16)
    for (int m = 0; m < 4; ++m) {
      bf16x16 aq = frag_a(s_q, m * 16, head * 32, DIMC, lane);
#pragma unroll
      for (int nt = 0; nt < 4; ++nt) {
        bf16x16 bk = frag_bt(s_k, head * 32, nt * 16, DIMC, lane);  // K^T via index swap
        f32x8 c = wmma_bf16(aq, bk, zero8());
#pragma unroll
        for (int r = 0; r < 8; ++r) {
          const int q = m * 16 + r + hi8;
          const int kk = nt * 16 + (lane & 15);
          float sval = -1e9f;  // padded rows/cols: finite -> softmax stays NaN-free
          if (q < 49 && kk < 49)
            sval = c[r] * ATT_SCALE + btab[relidx[q * 49 + kk] * NHEADS + head] +
                   mask[(w_in * 49 + q) * 49 + kk];
          sc[(r + hi8) * 64 + kk] = sval;
        }
      }
      // softmax over 64 (padded) keys, 16 rows; write bf16 probs over the score rows
      for (int r = 0; r < 16; ++r) {
        const float s0 = sc[r * 64 + lane];
        const float s1 = sc[r * 64 + 32 + lane];
        const float mx = wave_max(fmaxf(s0, s1));
        const float e0 = __expf(s0 - mx), e1 = __expf(s1 - mx);
        const float inv = 1.f / wave_sum(e0 + e1);
        psc[r * 128 + lane]      = f2bf(e0 * inv);
        psc[r * 128 + 32 + lane] = f2bf(e1 * inv);
      }
      // P @ V  (16x64 @ 64x32) -> attention output columns of this head
#pragma unroll
      for (int n2 = 0; n2 < 2; ++n2) {
        f32x8 acc = zero8();
#pragma unroll
        for (int kt = 0; kt < 2; ++kt) {
          bf16x16 ap = frag_a(psc, 0, kt * 32, 128, lane);
          bf16x16 bv = frag_bt(s_v, kt * 32, head * 32 + n2 * 16, 64, lane);  // V^T rows
          acc = wmma_bf16(ap, bv, acc);
        }
#pragma unroll
        for (int r = 0; r < 8; ++r)
          s_x[(m * 16 + r + hi8) * DIMC + head * 32 + n2 * 16 + (lane & 15)] = f2bf(acc[r]);
      }
    }
  }
  __syncthreads();

  // ---- Phase 3: projection + window-reverse + unshift + residual into x1 ----
  for (int t = wid; t < 64; t += 8) {
    const int m = t >> 4, n = t & 15;
    f32x8 acc = zero8();
#pragma unroll
    for (int kb = 0; kb < 8; ++kb) {
      bf16x16 a = frag_a(s_x, m * 16, kb * 32, DIMC, lane);
      bf16x16 b = frag_b_swz(wproj, kb * 16 + n, lane);
      acc = wmma_bf16(a, b, acc);
    }
    const int col = n * 16 + (lane & 15);
#pragma unroll
    for (int r = 0; r < 8; ++r) {
      const int rr = m * 16 + r + hi8;
      if (rr < 49) {
        const int i = rr / WS, j = rr % WS;
        const int ho = (wh * WS + i + SHIFT) % IH;
        const int wo = (ww * WS + j + SHIFT) % IW;
        const size_t gi = ((size_t)bimg * IH * IW + ho * IW + wo) * DIMC + col;
        x1[gi] = x[gi] + acc[r] + projb[col];
      }
    }
  }
}

// =====================================================================================
// Fused MLP kernel: LN2 + fc1 + exact GELU + fc2 + residual. One block per 64 tokens,
// 8 waves. Hidden activation streamed through a 4 KB LDS buffer in chunks of 32
// (never touches HBM). Wave (m, half) owns a 16x128 f32 output accumulator (64 VGPRs).
// =====================================================================================
__global__ __launch_bounds__(256) void swin_mlp_fused(
    const float* __restrict__ x1, const float* __restrict__ n2g, const float* __restrict__ n2b,
    const unsigned short* __restrict__ fc1w, const float* __restrict__ fc1b,
    const unsigned short* __restrict__ fc2w, const float* __restrict__ fc2b,
    float* __restrict__ out) {
  __shared__ unsigned short s_a[64 * 256];  // LN2(x1) in bf16
  __shared__ unsigned short s_h[64 * 32];   // gelu(fc1) hidden chunk in bf16

  const int tid = threadIdx.x, lane = tid & 31, wid = tid >> 5;
  const int row0 = blockIdx.x * 64;
  const int hi8 = (lane >= 16) ? 8 : 0;

  for (int r = wid; r < 64; r += 8) {
    const float* src = x1 + (size_t)(row0 + r) * DIMC;
    const f32x4 v0 = ((const f32x4*)(src + lane * 8))[0];
    const f32x4 v1 = ((const f32x4*)(src + lane * 8))[1];
    float v[8] = {v0.x, v0.y, v0.z, v0.w, v1.x, v1.y, v1.z, v1.w};
    float s = 0.f, ss = 0.f;
#pragma unroll
    for (int u = 0; u < 8; ++u) { s += v[u]; ss += v[u] * v[u]; }
    s = wave_sum(s); ss = wave_sum(ss);
    const float mu = s * (1.f / DIMC);
    const float rs = rsqrtf(ss * (1.f / DIMC) - mu * mu + LNEPS);
#pragma unroll
    for (int u = 0; u < 8; ++u) {
      const int c = lane * 8 + u;
      s_a[r * DIMC + c] = f2bf((v[u] - mu) * rs * n2g[c] + n2b[c]);
    }
  }
  __syncthreads();

  const int m = wid >> 1, hf = wid & 1;
  f32x8 acc[8];
#pragma unroll
  for (int t = 0; t < 8; ++t) acc[t] = zero8();

  for (int chunk = 0; chunk < HID / 32; ++chunk) {
    // fc1 tile (16x16, K=256) + bias + exact GELU -> bf16 LDS
    f32x8 h = zero8();
#pragma unroll
    for (int kb = 0; kb < 8; ++kb) {
      bf16x16 a = frag_a(s_a, m * 16, kb * 32, DIMC, lane);
      bf16x16 b = frag_b_swz(fc1w, kb * 64 + chunk * 2 + hf, lane);  // N=1024 -> 64 tiles/Kb
      h = wmma_bf16(a, b, h);
    }
#pragma unroll
    for (int r = 0; r < 8; ++r) {
      const int rr = m * 16 + r + hi8;
      const int cl = hf * 16 + (lane & 15);
      float vv = h[r] + fc1b[chunk * 32 + cl];
      vv = 0.5f * vv * (1.f + erff(vv * 0.7071067811865476f));  // exact GELU
      s_h[rr * 32 + cl] = f2bf(vv);
    }
    __syncthreads();
    // fc2 accumulation: A = hidden chunk (16x32), 8 output N-tiles per wave
    bf16x16 a2 = frag_a(s_h, m * 16, 0, 32, lane);
#pragma unroll
    for (int t = 0; t < 8; ++t) {
      bf16x16 b = frag_b_swz(fc2w, chunk * 16 + hf * 8 + t, lane);  // N=256 -> 16 tiles/Kb
      acc[t] = wmma_bf16(a2, b, acc[t]);
    }
    __syncthreads();  // before next chunk overwrites s_h
  }

#pragma unroll
  for (int t = 0; t < 8; ++t) {
    const int col = hf * 128 + t * 16 + (lane & 15);
#pragma unroll
    for (int r = 0; r < 8; ++r) {
      const size_t gi = (size_t)(row0 + m * 16 + r + hi8) * DIMC + col;
      out[gi] = x1[gi] + acc[t][r] + fc2b[col];
    }
  }
}

extern "C" void kernel_launch(void* const* d_in, const int* in_sizes, int n_in,
                              void* d_out, int out_size, void* d_ws, size_t ws_size,
                              hipStream_t stream) {
  (void)in_sizes; (void)n_in; (void)out_size; (void)ws_size;
  const float* x    = (const float*)d_in[0];
  const float* mask = (const float*)d_in[1];
  const float* n1g  = (const float*)d_in[2];
  const float* n1b  = (const float*)d_in[3];
  const float* wq_f = (const float*)d_in[4];
  const float* wk_f = (const float*)d_in[5];
  const float* wv_f = (const float*)d_in[6];
  const float* btab = (const float*)d_in[7];
  const float* wp_f = (const float*)d_in[8];
  const float* pb   = (const float*)d_in[9];
  const float* n2g  = (const float*)d_in[10];
  const float* n2b  = (const float*)d_in[11];
  const float* f1_f = (const float*)d_in[12];
  const float* f1b  = (const float*)d_in[13];
  const float* f2_f = (const float*)d_in[14];
  const float* f2b  = (const float*)d_in[15];
  const int* relidx = (const int*)d_in[16];

  // workspace layout
  char* wsb = (char*)d_ws;
  float* x1 = (float*)wsb;
  size_t off = (size_t)BATCH * IH * IW * DIMC * sizeof(float);  // 102,760,448 B
  unsigned short* wqb = (unsigned short*)(wsb + off); off += (size_t)DIMC * DIMC * 2;
  unsigned short* wkb = (unsigned short*)(wsb + off); off += (size_t)DIMC * DIMC * 2;
  unsigned short* wvb = (unsigned short*)(wsb + off); off += (size_t)DIMC * DIMC * 2;
  unsigned short* wpb = (unsigned short*)(wsb + off); off += (size_t)DIMC * DIMC * 2;
  unsigned short* f1b16 = (unsigned short*)(wsb + off); off += (size_t)DIMC * HID * 2;
  unsigned short* f2b16 = (unsigned short*)(wsb + off); off += (size_t)HID * DIMC * 2;

  const int nsq = DIMC * DIMC;   // 65536
  const int nfc = DIMC * HID;    // 262144
  cvt_swz<<<(nsq + 255) / 256, 256, 0, stream>>>(wq_f, wqb, DIMC, DIMC);
  cvt_swz<<<(nsq + 255) / 256, 256, 0, stream>>>(wk_f, wkb, DIMC, DIMC);
  cvt_swz<<<(nsq + 255) / 256, 256, 0, stream>>>(wv_f, wvb, DIMC, DIMC);
  cvt_swz<<<(nsq + 255) / 256, 256, 0, stream>>>(wp_f, wpb, DIMC, DIMC);
  cvt_swz<<<(nfc + 255) / 256, 256, 0, stream>>>(f1_f, f1b16, DIMC, HID);
  cvt_swz<<<(nfc + 255) / 256, 256, 0, stream>>>(f2_f, f2b16, HID, DIMC);

  swin_attn_fused<<<BATCH * NWIN, 256, 0, stream>>>(
      x, mask, n1g, n1b, wqb, wkb, wvb, btab, relidx, wpb, pb, x1);

  swin_mlp_fused<<<(BATCH * IH * IW) / 64, 256, 0, stream>>>(
      x1, n2g, n2b, f1b16, f1b, f2b16, f2b, (float*)d_out);
}
